// match_block_22419729285784
// MI455X (gfx1250) — matrix operand order repaired and checked by reading, not compiled, not run on hardware
//
#include <hip/hip_runtime.h>

typedef __attribute__((ext_vector_type(16))) _Float16 v16h;
typedef __attribute__((ext_vector_type(8)))  _Float16 v8h;
typedef __attribute__((ext_vector_type(8)))  float    v8f;

#define C_DIM 256
#define I_DIM 128
#define NA    4096
#define ND    16384
#define EPSV  1e-5f

// Fragment-order LDS layout: slot = tile*32 + lane, 16 contiguous halves/slot,
// pitch 24 halves (48 B -> every slot 16-B aligned, lane stride 12 dwords).
#define APITCH 24
#define ADW    12
#define FRAG_HALVES(T) ((T) * 2 * APITCH)   // T rows/cols of 16 -> (T/16)*32*APITCH

// ---------------------------------------------------------------------------
// WMMA wrapper (wave32, V_WMMA_F32_16X16X32_F16)
// ---------------------------------------------------------------------------
__device__ __forceinline__ v8f wmma(v16h a, v16h b, v8f c) {
  return __builtin_amdgcn_wmma_f32_16x16x32_f16(
      /*neg_a=*/false, a, /*neg_b=*/false, b,
      /*c_mod=*/(short)0, c, /*reuse_a=*/false, /*reuse_b=*/false);
}

// Two ds_load_b128 per fragment.
__device__ __forceinline__ v16h load_frag(const _Float16* lds, int slot) {
  const v8h* p = (const v8h*)(lds + slot * APITCH);
  v8h lo = p[0], hi = p[1];
  return __builtin_shufflevector(lo, hi, 0, 1, 2, 3, 4, 5, 6, 7, 8, 9, 10, 11,
                                 12, 13, 14, 15);
}

__device__ __forceinline__ unsigned pack2(float a, float b) {
  union { _Float16 h[2]; unsigned u; } v;
  v.h[0] = (_Float16)a; v.h[1] = (_Float16)b;
  return v.u;
}
__device__ __forceinline__ unsigned pack2h(_Float16 a, _Float16 b) {
  union { _Float16 h[2]; unsigned u; } v;
  v.h[0] = a; v.h[1] = b;
  return v.u;
}

// ---------------------------------------------------------------------------
// Staging: global -> fragment-order LDS. A-frag (16x32): lane<16 holds row m,
// K {0..7,16..23}; lane>=16 holds K {8..15,24..31}; half pair (2j,2j+1) is
// k-contiguous. B-frag (32x16): same K order as A-frag.
// ---------------------------------------------------------------------------
template <int BM>
__device__ __forceinline__ void stage_a_f16(const _Float16* __restrict__ g,
                                            int ld, int m0, int k0,
                                            _Float16* lds, int tid) {
  unsigned* out = (unsigned*)lds;
#pragma unroll
  for (int t = 0; t < BM * 16 / 256; ++t) {
    int d = tid + t * 256;
    int slot = d >> 3, j = d & 7;
    int lane = slot & 31, mt = slot >> 5;
    int m = m0 + mt * 16 + (lane & 15);
    int k = k0 + ((lane < 16) ? 0 : 8) + ((2 * j) & 7) + ((j >= 4) ? 16 : 0);
    out[slot * ADW + j] = *(const unsigned*)(g + (size_t)m * ld + k);
  }
}

template <int BM>
__device__ __forceinline__ void stage_a_f32(const float* __restrict__ g, int ld,
                                            int m0, int k0, _Float16* lds,
                                            int tid) {
  unsigned* out = (unsigned*)lds;
#pragma unroll
  for (int t = 0; t < BM * 16 / 256; ++t) {
    int d = tid + t * 256;
    int slot = d >> 3, j = d & 7;
    int lane = slot & 31, mt = slot >> 5;
    int m = m0 + mt * 16 + (lane & 15);
    int k = k0 + ((lane < 16) ? 0 : 8) + ((2 * j) & 7) + ((j >= 4) ? 16 : 0);
    const float* s = g + (size_t)m * ld + k;
    out[slot * ADW + j] = pack2(s[0], s[1]);
  }
}

template <int BN>
__device__ __forceinline__ void stage_b_f32(const float* __restrict__ g, int ld,
                                            int k0, int n0, _Float16* lds,
                                            int tid) {
  unsigned* out = (unsigned*)lds;
#pragma unroll
  for (int t = 0; t < BN * 16 / 256; ++t) {
    int d = tid + t * 256;
    int slot = d >> 3, j = d & 7;
    int lane = slot & 31, nt = slot >> 5;
    int n = n0 + nt * 16 + (lane & 15);
    int k = k0 + ((lane < 16) ? 0 : 8) + ((2 * j) & 7) + ((j >= 4) ? 16 : 0);
    out[slot * ADW + j] =
        pack2(g[(size_t)k * ld + n], g[(size_t)(k + 1) * ld + n]);
  }
}

template <int BN>
__device__ __forceinline__ void stage_b_f16(const _Float16* __restrict__ g,
                                            int ld, int k0, int n0,
                                            _Float16* lds, int tid) {
  unsigned* out = (unsigned*)lds;
#pragma unroll
  for (int t = 0; t < BN * 16 / 256; ++t) {
    int d = tid + t * 256;
    int slot = d >> 3, j = d & 7;
    int lane = slot & 31, nt = slot >> 5;
    int n = n0 + nt * 16 + (lane & 15);
    int k = k0 + ((lane < 16) ? 0 : 8) + ((2 * j) & 7) + ((j >= 4) ? 16 : 0);
    out[slot * ADW + j] =
        pack2h(g[(size_t)k * ld + n], g[(size_t)(k + 1) * ld + n]);
  }
}

// ---------------------------------------------------------------------------
// Kernel 1: a_g = g_w@aim + g_b (I x Na), thT = (theta_w@aim + theta_b)^T
// BM = 128 (= I), BN = 64, K = C = 256. 8 waves, one 16-row strip each.
// ---------------------------------------------------------------------------
__global__ __launch_bounds__(256) void k_proj_aim(
    const float* __restrict__ aim, const float* __restrict__ g_w,
    const float* __restrict__ g_b, const float* __restrict__ th_w,
    const float* __restrict__ th_b, _Float16* __restrict__ ag,
    _Float16* __restrict__ thT) {
  const int b = blockIdx.y;
  const int n0 = blockIdx.x * 64;
  const int tid = threadIdx.x;
  const int lane = tid & 31;
  const int w = tid >> 5;

  __shared__ alignas(16) _Float16 Afg[FRAG_HALVES(128)];
  __shared__ alignas(16) _Float16 Aft[FRAG_HALVES(128)];
  __shared__ alignas(16) _Float16 Bf[FRAG_HALVES(64)];

  const float* aimB = aim + (size_t)b * C_DIM * NA;

  v8f accg[4] = {};
  v8f acct[4] = {};

  for (int k0 = 0; k0 < C_DIM; k0 += 32) {
    stage_a_f32<128>(g_w, C_DIM, 0, k0, Afg, tid);
    stage_a_f32<128>(th_w, C_DIM, 0, k0, Aft, tid);
    stage_b_f32<64>(aimB, NA, k0, n0, Bf, tid);
    __syncthreads();
    v16h fg = load_frag(Afg, w * 32 + lane);
    v16h ft = load_frag(Aft, w * 32 + lane);
#pragma unroll
    for (int nt = 0; nt < 4; ++nt) {
      v16h fb = load_frag(Bf, nt * 32 + lane);
      accg[nt] = wmma(fg, fb, accg[nt]);
      acct[nt] = wmma(ft, fb, acct[nt]);
    }
    __syncthreads();
  }

  const int mrow = w * 16 + 8 * (lane >> 4);
  const int ncol = lane & 15;
  _Float16* agB = ag + (size_t)b * I_DIM * NA;
  _Float16* thB = thT + (size_t)b * NA * I_DIM;
#pragma unroll
  for (int nt = 0; nt < 4; ++nt) {
#pragma unroll
    for (int r = 0; r < 8; ++r) {
      int i = mrow + r;
      int a = n0 + nt * 16 + ncol;
      agB[(size_t)i * NA + a] = (_Float16)(accg[nt][r] + g_b[i]);
      thB[(size_t)a * I_DIM + i] = (_Float16)(acct[nt][r] + th_b[i]);
    }
  }
}

// ---------------------------------------------------------------------------
// Kernel 2: M = (a_g @ th_a^T) / Na -> (I x I) f16. One block per batch.
// BM = 128, BN = 128, K = Na = 4096.
// ---------------------------------------------------------------------------
__global__ __launch_bounds__(256) void k_gram(const _Float16* __restrict__ ag,
                                              const _Float16* __restrict__ thT,
                                              _Float16* __restrict__ Mm) {
  const int b = blockIdx.x;
  const int tid = threadIdx.x;
  const int lane = tid & 31;
  const int w = tid >> 5;

  __shared__ alignas(16) _Float16 Af[FRAG_HALVES(128)];
  __shared__ alignas(16) _Float16 Bf[FRAG_HALVES(128)];

  const _Float16* agB = ag + (size_t)b * I_DIM * NA;
  const _Float16* thB = thT + (size_t)b * NA * I_DIM;

  v8f acc[8] = {};
  for (int k0 = 0; k0 < NA; k0 += 32) {
    stage_a_f16<128>(agB, NA, 0, k0, Af, tid);
    stage_b_f16<128>(thB, I_DIM, k0, 0, Bf, tid);
    __syncthreads();
    v16h fa = load_frag(Af, w * 32 + lane);
#pragma unroll
    for (int nt = 0; nt < 8; ++nt) {
      v16h fb = load_frag(Bf, nt * 32 + lane);
      acc[nt] = wmma(fa, fb, acc[nt]);
    }
    __syncthreads();
  }

  const float inv_n = 1.0f / (float)NA;
  const int mrow = w * 16 + 8 * (lane >> 4);
  const int ncol = lane & 15;
  _Float16* MmB = Mm + (size_t)b * I_DIM * I_DIM;
#pragma unroll
  for (int nt = 0; nt < 8; ++nt)
#pragma unroll
    for (int r = 0; r < 8; ++r)
      MmB[(size_t)(mrow + r) * I_DIM + nt * 16 + ncol] =
          (_Float16)(acc[nt][r] * inv_n);
}

// ---------------------------------------------------------------------------
// Kernel 3a: W2 = q_w @ Mscaled -> (C x I) f16. Grid (2 mblocks, B).
// ---------------------------------------------------------------------------
__global__ __launch_bounds__(256) void k_w2(const float* __restrict__ q_w,
                                            const _Float16* __restrict__ Mm,
                                            _Float16* __restrict__ W2) {
  const int b = blockIdx.y;
  const int m0 = blockIdx.x * 128;
  const int tid = threadIdx.x;
  const int lane = tid & 31;
  const int w = tid >> 5;

  __shared__ alignas(16) _Float16 Af[FRAG_HALVES(128)];
  __shared__ alignas(16) _Float16 Bf[FRAG_HALVES(128)];

  const _Float16* MmB = Mm + (size_t)b * I_DIM * I_DIM;

  v8f acc[8] = {};
  for (int k0 = 0; k0 < I_DIM; k0 += 32) {
    stage_a_f32<128>(q_w, I_DIM, m0, k0, Af, tid);
    stage_b_f16<128>(MmB, I_DIM, k0, 0, Bf, tid);
    __syncthreads();
    v16h fa = load_frag(Af, w * 32 + lane);
#pragma unroll
    for (int nt = 0; nt < 8; ++nt) {
      v16h fb = load_frag(Bf, nt * 32 + lane);
      acc[nt] = wmma(fa, fb, acc[nt]);
    }
    __syncthreads();
  }

  const int mrow = m0 + w * 16 + 8 * (lane >> 4);
  const int ncol = lane & 15;
  _Float16* W2B = W2 + (size_t)b * C_DIM * I_DIM;
#pragma unroll
  for (int nt = 0; nt < 8; ++nt)
#pragma unroll
    for (int r = 0; r < 8; ++r)
      W2B[(size_t)(mrow + r) * I_DIM + nt * 16 + ncol] = (_Float16)acc[nt][r];
}

// ---------------------------------------------------------------------------
// Kernel 3b: W3 = s[c] * (W2 @ phi_w) -> (C x C) f16, BN scale folded in.
// Grid (4 nblocks, 2 mblocks, B). BM=128, BN=64, K=128.
// ---------------------------------------------------------------------------
__global__ __launch_bounds__(256) void k_w3(const _Float16* __restrict__ W2,
                                            const float* __restrict__ phi_w,
                                            const float* __restrict__ bn_gamma,
                                            const float* __restrict__ bn_var,
                                            _Float16* __restrict__ W3) {
  const int b = blockIdx.z;
  const int m0 = blockIdx.y * 128;
  const int n0 = blockIdx.x * 64;
  const int tid = threadIdx.x;
  const int lane = tid & 31;
  const int w = tid >> 5;

  __shared__ alignas(16) _Float16 Af[FRAG_HALVES(128)];
  __shared__ alignas(16) _Float16 Bf[FRAG_HALVES(64)];

  const _Float16* W2B = W2 + (size_t)b * C_DIM * I_DIM;

  v8f acc[4] = {};
  for (int k0 = 0; k0 < I_DIM; k0 += 32) {
    stage_a_f16<128>(W2B, I_DIM, m0, k0, Af, tid);
    stage_b_f32<64>(phi_w, C_DIM, k0, n0, Bf, tid);
    __syncthreads();
    v16h fa = load_frag(Af, w * 32 + lane);
#pragma unroll
    for (int nt = 0; nt < 4; ++nt) {
      v16h fb = load_frag(Bf, nt * 32 + lane);
      acc[nt] = wmma(fa, fb, acc[nt]);
    }
    __syncthreads();
  }

  const int mrow = m0 + w * 16 + 8 * (lane >> 4);
  const int ncol = lane & 15;
  _Float16* W3B = W3 + (size_t)b * C_DIM * C_DIM;
#pragma unroll
  for (int r = 0; r < 8; ++r) {
    int m = mrow + r;
    float s = bn_gamma[m] * rsqrtf(bn_var[m] + EPSV);
#pragma unroll
    for (int nt = 0; nt < 4; ++nt)
      W3B[(size_t)m * C_DIM + n0 + nt * 16 + ncol] = (_Float16)(acc[nt][r] * s);
  }
}

// ---------------------------------------------------------------------------
// Kernel 4: Btot[b,c] = s*(W2@phi_b + q_b) + (beta - mean*s)
// ---------------------------------------------------------------------------
__global__ __launch_bounds__(256) void k_bias(
    const _Float16* __restrict__ W2, const float* __restrict__ phi_b,
    const float* __restrict__ q_b, const float* __restrict__ bn_gamma,
    const float* __restrict__ bn_beta, const float* __restrict__ bn_mean,
    const float* __restrict__ bn_var, float* __restrict__ Btot) {
  const int b = blockIdx.x;
  const int c = threadIdx.x;
  const _Float16* row = W2 + ((size_t)b * C_DIM + c) * I_DIM;
  float sum = 0.f;
  for (int i = 0; i < I_DIM; ++i) sum += (float)row[i] * phi_b[i];
  float s = bn_gamma[c] * rsqrtf(bn_var[c] + EPSV);
  float t = bn_beta[c] - bn_mean[c] * s;
  Btot[b * C_DIM + c] = s * (sum + q_b[c]) + t;
}

// ---------------------------------------------------------------------------
// Kernel 5 (dominant): out = W3[b] @ detect[b] + Btot[b,c] + detect
// Grid (ND/64, 2, B). BM=128, BN=64, K=C=256, with next-tile prefetch.
// ---------------------------------------------------------------------------
__global__ __launch_bounds__(256) void k_final(const _Float16* __restrict__ W3,
                                               const float* __restrict__ detect,
                                               const float* __restrict__ Btot,
                                               float* __restrict__ out) {
  const int b = blockIdx.z;
  const int m0 = blockIdx.y * 128;
  const int n0 = blockIdx.x * 64;
  const int tid = threadIdx.x;
  const int lane = tid & 31;
  const int w = tid >> 5;

  __shared__ alignas(16) _Float16 Af[FRAG_HALVES(128)];
  __shared__ alignas(16) _Float16 Bf[FRAG_HALVES(64)];

  const _Float16* W3B = W3 + (size_t)b * C_DIM * C_DIM;
  const float* detB = detect + (size_t)b * C_DIM * ND;

  v8f acc[4] = {};
  for (int k0 = 0; k0 < C_DIM; k0 += 32) {
    if (k0 + 32 < C_DIM) {  // prefetch next K-tile (global_prefetch_b8)
      __builtin_prefetch(
          detB + (size_t)(k0 + 32 + (tid >> 3)) * ND + n0 + (tid & 7) * 8, 0, 1);
      __builtin_prefetch(
          W3B + (size_t)(m0 + (tid >> 1)) * C_DIM + k0 + 32 + (tid & 1) * 16, 0, 1);
    }
    stage_a_f16<128>(W3B, C_DIM, m0, k0, Af, tid);
    stage_b_f32<64>(detB, ND, k0, n0, Bf, tid);
    __syncthreads();
    v16h fa = load_frag(Af, w * 32 + lane);
#pragma unroll
    for (int nt = 0; nt < 4; ++nt) {
      v16h fb = load_frag(Bf, nt * 32 + lane);
      acc[nt] = wmma(fa, fb, acc[nt]);
    }
    __syncthreads();
  }

  const int mrow = m0 + w * 16 + 8 * (lane >> 4);
  const int ncol = lane & 15;
  float* outB = out + (size_t)b * C_DIM * ND;
#pragma unroll
  for (int r = 0; r < 8; ++r) {
    int m = mrow + r;
    float bb = Btot[b * C_DIM + m];
#pragma unroll
    for (int nt = 0; nt < 4; ++nt) {
      int col = n0 + nt * 16 + ncol;
      outB[(size_t)m * ND + col] = acc[nt][r] + bb + detB[(size_t)m * ND + col];
    }
  }
}

// ---------------------------------------------------------------------------
extern "C" void kernel_launch(void* const* d_in, const int* in_sizes, int n_in,
                              void* d_out, int out_size, void* d_ws,
                              size_t ws_size, hipStream_t stream) {
  const float* detect = (const float*)d_in[0];
  const float* aim    = (const float*)d_in[1];
  const float* g_w    = (const float*)d_in[2];
  const float* g_b    = (const float*)d_in[3];
  const float* th_w   = (const float*)d_in[4];
  const float* th_b   = (const float*)d_in[5];
  const float* phi_w  = (const float*)d_in[6];
  const float* phi_b  = (const float*)d_in[7];
  const float* q_w    = (const float*)d_in[8];
  const float* q_b    = (const float*)d_in[9];
  const float* gamma_ = (const float*)d_in[10];
  const float* beta_  = (const float*)d_in[11];
  const float* mean_  = (const float*)d_in[12];
  const float* var_   = (const float*)d_in[13];
  float* out = (float*)d_out;

  char* ws = (char*)d_ws;
  _Float16* ag   = (_Float16*)(ws);                              // 4 MiB
  _Float16* thT  = (_Float16*)(ws + (4ull << 20));               // 4 MiB
  _Float16* Mm   = (_Float16*)(ws + (8ull << 20));               // 128 KiB
  _Float16* W2   = (_Float16*)(ws + (8ull << 20) + (256 << 10)); // 256 KiB
  _Float16* W3   = (_Float16*)(ws + (9ull << 20));               // 512 KiB
  float*    Btot = (float*)(ws + (9ull << 20) + (768 << 10));    // 4 KiB

  const int B = 4;
  k_proj_aim<<<dim3(NA / 64, B), 256, 0, stream>>>(aim, g_w, g_b, th_w, th_b,
                                                   ag, thT);
  k_gram<<<dim3(B), 256, 0, stream>>>(ag, thT, Mm);
  k_w2<<<dim3(2, B), 256, 0, stream>>>(q_w, Mm, W2);
  k_w3<<<dim3(4, 2, B), 256, 0, stream>>>(W2, phi_w, gamma_, var_, W3);
  k_bias<<<dim3(B), 256, 0, stream>>>(W2, phi_b, q_b, gamma_, beta_, mean_,
                                      var_, Btot);
  k_final<<<dim3(ND / 64, 2, B), 256, 0, stream>>>(W3, detect, Btot, out);
}
